// SumLayer_82394652606955
// MI455X (gfx1250) — matrix operand order repaired and checked by reading, not compiled, not run on hardware
//
#include <hip/hip_runtime.h>

// ---------------------------------------------------------------------------
// SumLayer: out[s,n] = log( sum_k exp(ll[s,k] + w[n,k]) ) - logsumexp_k w[n,k]
//          = log( exp(ll) @ exp(w)^T ) - log_z            (K = C1 + C2 = 4096)
//
// Strategy (MI455X / gfx1250, wave32):
//  * Compute-bound (≈344 FLOP/byte) -> use v_wmma_f32_16x16x32_f16.
//  * exp(N(0,1)) fits f16 range; accumulate in f32 -> ~1e-3 abs error in log.
//  * expA (32MB f16) + expB (8MB f16) fit in 192MB L2 -> direct global loads,
//    64x64 register tile per wave (16 WMMA accumulators) for 4x L2-traffic cut.
// ---------------------------------------------------------------------------

typedef _Float16 v16h __attribute__((ext_vector_type(16)));
typedef _Float16 v8h  __attribute__((ext_vector_type(8)));
typedef _Float16 v4h  __attribute__((ext_vector_type(4)));
typedef float    v8f  __attribute__((ext_vector_type(8)));
typedef float    v4f  __attribute__((ext_vector_type(4)));

#define SS 4096   // samples
#define CC 2048   // children per layer
#define KK 4096   // concatenated K = 2*CC
#define NN 1024   // nodes

// ---- Phase 1: dst[r, k] = (f16) exp( k<CC ? x1[r,k] : x2[r,k-CC] ) ---------
// 4 contiguous elements per thread; grid exactly covers rows*4096 elements.
__global__ void __launch_bounds__(256)
build_exp_rows(const float* __restrict__ x1, const float* __restrict__ x2,
               _Float16* __restrict__ dst) {
  size_t idx = ((size_t)blockIdx.x * blockDim.x + threadIdx.x) * 4;
  size_t row = idx >> 12;          // /4096
  int    k   = (int)(idx & 4095);
  const float* src = (k < CC) ? (x1 + row * CC + k)
                              : (x2 + row * CC + (k - CC));
  v4f v = *(const v4f*)src;
  v4h h;
  h[0] = (_Float16)__expf(v[0]);
  h[1] = (_Float16)__expf(v[1]);
  h[2] = (_Float16)__expf(v[2]);
  h[3] = (_Float16)__expf(v[3]);
  *(v4h*)(dst + idx) = h;
}

// ---- Phase 2: log_z[n] = log( sum_k exp(w1[n,:]) + exp(w2[n,:]) ) ----------
__global__ void __launch_bounds__(256)
node_logz(const float* __restrict__ w1, const float* __restrict__ w2,
          float* __restrict__ logz) {
  __shared__ float red[256];
  const int n = blockIdx.x;
  const int t = threadIdx.x;
  float s = 0.0f;
  for (int k = t * 4; k < CC; k += 256 * 4) {
    v4f a = *(const v4f*)(w1 + (size_t)n * CC + k);
    v4f b = *(const v4f*)(w2 + (size_t)n * CC + k);
    s += __expf(a[0]) + __expf(a[1]) + __expf(a[2]) + __expf(a[3]);
    s += __expf(b[0]) + __expf(b[1]) + __expf(b[2]) + __expf(b[3]);
  }
  red[t] = s;
  __syncthreads();
  for (int off = 128; off > 0; off >>= 1) {
    if (t < off) red[t] += red[t + off];
    __syncthreads();
  }
  if (t == 0) logz[n] = __logf(red[0]);
}

// ---- Phase 3: WMMA GEMM + fused log/normalize epilogue ---------------------
// One wave (32 threads) per 64x64 output tile; 4x4 grid of 16x16 WMMA tiles.
// A: [SS][KK] f16 row-major, B: [NN][KK] f16 row-major (i.e. B^T operand).
__global__ void __launch_bounds__(32)
gemm_log(const _Float16* __restrict__ A, const _Float16* __restrict__ B,
         const float* __restrict__ logz, float* __restrict__ out) {
  const int lane = threadIdx.x & 31;
  const int r    = lane & 15;   // row-in-tile (A) / col-in-tile (B,C)
  const int half = lane >> 4;   // K-group selector per ISA layout
  const int bm   = blockIdx.x * 64;   // sample base
  const int bn   = blockIdx.y * 64;   // node base

  v8f acc[4][4] = {};

  const _Float16* arow[4];
  const _Float16* brow[4];
#pragma unroll
  for (int t = 0; t < 4; ++t) {
    arow[t] = A + (size_t)(bm + t * 16 + r) * KK;
    brow[t] = B + (size_t)(bn + t * 16 + r) * KK;
  }

  for (int k0 = 0; k0 < KK; k0 += 32) {
    v16h a[4], b[4];
#pragma unroll
    for (int t = 0; t < 4; ++t) {
      // A 16x32 f16 layout: v0..3 = K(8h..8h+7), v4..7 = K(16+8h..16+8h+7)
      v8h lo = *(const v8h*)(arow[t] + k0 + half * 8);
      v8h hi = *(const v8h*)(arow[t] + k0 + 16 + half * 8);
      a[t] = __builtin_shufflevector(lo, hi, 0, 1, 2, 3, 4, 5, 6, 7,
                                             8, 9, 10, 11, 12, 13, 14, 15);
      // B 32x16 f16 layout: lane=col n, v0..3 = K(8h..8h+7), v4..7 = K(16+8h..16+8h+7)
      v8h blo = *(const v8h*)(brow[t] + k0 + half * 8);
      v8h bhi = *(const v8h*)(brow[t] + k0 + 16 + half * 8);
      b[t] = __builtin_shufflevector(blo, bhi, 0, 1, 2, 3, 4, 5, 6, 7,
                                               8, 9, 10, 11, 12, 13, 14, 15);
    }
#pragma unroll
    for (int i = 0; i < 4; ++i)
#pragma unroll
      for (int j = 0; j < 4; ++j)
        acc[i][j] = __builtin_amdgcn_wmma_f32_16x16x32_f16(
            /*neg_a=*/false, a[i], /*neg_b=*/false, b[j],
            /*c_mod=*/(short)0, acc[i][j],
            /*reuse_a=*/false, /*reuse_b=*/false);
  }

  // C/D layout: col = tile_col + (lane&15); row = tile_row + 8*half + vgpr
#pragma unroll
  for (int i = 0; i < 4; ++i) {
#pragma unroll
    for (int j = 0; j < 4; ++j) {
      const int col = bn + j * 16 + r;
      const float lz = logz[col];
#pragma unroll
      for (int e = 0; e < 8; ++e) {
        const int row = bm + i * 16 + half * 8 + e;
        out[(size_t)row * NN + col] = __logf(acc[i][j][e]) - lz;
      }
    }
  }
}

// ---------------------------------------------------------------------------
extern "C" void kernel_launch(void* const* d_in, const int* in_sizes, int n_in,
                              void* d_out, int out_size, void* d_ws, size_t ws_size,
                              hipStream_t stream) {
  (void)in_sizes; (void)n_in; (void)out_size; (void)ws_size;
  const float* ll1 = (const float*)d_in[0];  // (S, C1) f32
  const float* ll2 = (const float*)d_in[1];  // (S, C2) f32
  const float* w1  = (const float*)d_in[2];  // (N, C1) f32
  const float* w2  = (const float*)d_in[3];  // (N, C2) f32
  float* out = (float*)d_out;                // (S, N)  f32

  // Workspace layout: expA f16 [SS][KK] | expB f16 [NN][KK] | logz f32 [NN]
  _Float16* expA = (_Float16*)d_ws;
  _Float16* expB = expA + (size_t)SS * KK;
  float*    logz = (float*)(expB + (size_t)NN * KK);

  // Phase 1: fused concat + exp + f32->f16
  build_exp_rows<<<(SS * (size_t)KK / 4) / 256, 256, 0, stream>>>(ll1, ll2, expA);
  build_exp_rows<<<(NN * (size_t)KK / 4) / 256, 256, 0, stream>>>(w1, w2, expB);

  // Phase 2: per-node normalization constant
  node_logz<<<NN, 256, 0, stream>>>(w1, w2, logz);

  // Phase 3: WMMA GEMM (4096x1024x4096) with fused log - log_z epilogue
  gemm_log<<<dim3(SS / 64, NN / 64), 32, 0, stream>>>(expA, expB, logz, out);
}
